// HeteroGraphEncoder_11252814315838
// MI455X (gfx1250) — matrix operand order repaired and checked by reading, not compiled, not run on hardware
//
#include <hip/hip_runtime.h>
#include <hip/hip_bf16.h>

// ---------------------------------------------------------------------------
// HeteroGraphEncoder for MI455X (gfx1250, wave32, WMMA)
//
// GEMMs via v_wmma_f32_16x16x32_bf16 (fp32 accumulate), edge aggregation via
// global_atomic_add_f32 (HBM/L2 bound, ~2.5GB -> ~107us floor at 23.3 TB/s).
//
// GEMM: 128row x 256col tile per 256-thread block (8 waves x 16 WMMA tiles);
// A streamed from HBM exactly once. W pre-packed to bf16 in WMMA B-fragment
// order (<=256KB, L2-resident) so per-K-step B staging is a raw 16KB
// global->LDS copy. Both A and B tiles double-buffered in LDS with an
// 80B-per-row stride (bank-conflict-free fragment reads). B fragments are
// software-pipelined one WMMA tile ahead so ds loads overlap the matrix pipe.
// ---------------------------------------------------------------------------

typedef __attribute__((ext_vector_type(16))) __bf16 v16bf;
typedef __attribute__((ext_vector_type(8)))  __bf16 v8bf;
typedef __attribute__((ext_vector_type(4)))  __bf16 v4bf;
typedef __attribute__((ext_vector_type(8)))  float  v8f;
typedef __attribute__((ext_vector_type(4)))  float  v4f;

#define H_DIM 256
#define LSTR  40   // LDS row/col stride in bf16 elements (80B): conflict-free

// Pack W[K x 256] fp32 -> bf16 in fragment order: out[((kt*256)+c)*32 + kk]
// = W[(kt*32+kk)*256 + c].
__global__ __launch_bounds__(256)
void hge_pack_w(const float* __restrict__ W, __bf16* __restrict__ out, int K) {
  int o = blockIdx.x * 256 + threadIdx.x;      // 0 .. K*256-1
  if (o >= K * H_DIM) return;
  int kk = o & 31;
  int c  = (o >> 5) & 255;
  int kt = o >> 13;                            // / (256*32)
  out[o] = (__bf16)W[(size_t)(kt * 32 + kk) * H_DIM + c];
}

// Load a 32B fragment as two 16B halves (LSTR*2 is 16B- but not 32B-aligned).
__device__ __forceinline__ v16bf hge_ld_frag(const __bf16* p) {
  union { v16bf v; v8bf h[2]; } u;
  u.h[0] = *(const v8bf*)p;
  u.h[1] = *(const v8bf*)(p + 8);
  return u.v;
}

// Stage one 128x32 A tile (fp32 -> bf16, contiguous K order).
__device__ __forceinline__ void hge_stage_a(
    const float* __restrict__ A, __bf16* __restrict__ sA,
    int rowBase, int k0, int M, int K, int tid) {
#pragma unroll
  for (int it = 0; it < 4; ++it) {
    int idx = tid + it * 256;              // 0..1023
    int r   = idx >> 3;                    // tile row 0..127
    int kq  = idx & 7;                     // float4 chunk within K32
    int gr  = rowBase + r;
    int cr  = (gr < M) ? gr : (M - 1);     // clamp: load always valid
    v4f v = *(const v4f*)(A + (size_t)cr * K + (k0 + (kq << 2)));
    bool ok = (gr < M);
    v4bf b;
#pragma unroll
    for (int j = 0; j < 4; ++j) b[j] = (__bf16)(ok ? v[j] : 0.f);
    int jq = kq;                                     // contiguous k-groups
    *(v4bf*)(sA + r * LSTR + (jq << 2)) = b;
  }
}

// Stage one 256x32 B ktile: contiguous 16KB copy from packed W into LDS.
__device__ __forceinline__ void hge_stage_b(
    const __bf16* __restrict__ bp, __bf16* __restrict__ sB, int tid) {
#pragma unroll
  for (int it = 0; it < 4; ++it) {
    int idx = tid + it * 256;              // 0..1023 chunks of 8 bf16
    int c   = idx >> 2;                    // col 0..255
    int ch  = idx & 3;                     // 8-elem chunk within col
    v8bf v = *(const v8bf*)(bp + (c << 5) + (ch << 3));
    *(v8bf*)(sB + c * LSTR + (ch << 3)) = v;
  }
}

// C[M x 256] = A[M x K] * Wpk + bias   (K % 32 == 0, Wpk = packed bf16 W)
// block = 256 threads (8 waves); tile = 128 rows x 256 cols; K step = 32.
__global__ __launch_bounds__(256)
void hge_gemm_bias(const float* __restrict__ A, const __bf16* __restrict__ Wpk,
                   const float* __restrict__ bias, float* __restrict__ C,
                   int M, int K) {
  __shared__ __align__(16) __bf16 ldsA[2][128 * LSTR];   // 2 x 10KB
  __shared__ __align__(16) __bf16 ldsB[2][256 * LSTR];   // 2 x 20KB

  const int tid  = threadIdx.x;
  const int wv   = tid >> 5;        // wave id 0..7 -> rows wv*16..wv*16+15
  const int lane = tid & 31;
  const int ln   = lane & 15;
  const int kh   = lane >> 4;
  const int rowBase = blockIdx.x * 128;

  v8f acc[16];
#pragma unroll
  for (int t = 0; t < 16; ++t)
    acc[t] = (v8f){0.f, 0.f, 0.f, 0.f, 0.f, 0.f, 0.f, 0.f};

  const int aOff = (wv * 16 + ln) * LSTR + (kh << 4);
  const int bOff = ln * LSTR + (kh << 4);

  hge_stage_a(A, ldsA[0], rowBase, 0, M, K, tid);
  hge_stage_b(Wpk, ldsB[0], tid);

  int buf = 0;
  for (int k0 = 0; k0 < K; k0 += 32, buf ^= 1) {
    __syncthreads();                       // stage(buf) complete
    if (k0 + 32 < K) {                     // prefetch next tiles (uniform)
      hge_stage_a(A, ldsA[buf ^ 1], rowBase, k0 + 32, M, K, tid);
      hge_stage_b(Wpk + ((size_t)((k0 >> 5) + 1) << 13), ldsB[buf ^ 1], tid);
    }

    const __bf16* sB = ldsB[buf] + bOff;   // fragment t at sB + t*16*LSTR
    v16bf afrag = hge_ld_frag(ldsA[buf] + aOff);
    v16bf bcur  = hge_ld_frag(sB);
#pragma unroll
    for (int t = 0; t < 16; ++t) {
      v16bf bnext = bcur;                  // software pipeline: load t+1
      if (t < 15) bnext = hge_ld_frag(sB + (t + 1) * (16 * LSTR));
      acc[t] = __builtin_amdgcn_wmma_f32_16x16x32_bf16(
          /*neg_a=*/false, afrag, /*neg_b=*/false, bcur,
          /*c_mod=*/(short)0, acc[t], /*reuse_a=*/false, /*reuse_b=*/false);
      bcur = bnext;
    }
  }

  // ---- store with fused bias. C/D layout: VGPR r -> M = r + 8*(lane>=16) ----
  const int lm = kh << 3;
#pragma unroll
  for (int t = 0; t < 16; ++t) {
    int gc = t * 16 + ln;
    float bv = bias[gc];
#pragma unroll
    for (int r = 0; r < 8; ++r) {
      int gr = rowBase + wv * 16 + lm + r;
      if (gr < M) C[(size_t)gr * H_DIM + gc] = acc[t][r] + bv;
    }
  }
}

// out[dst[e]] += msg[src[e]] * w[e]; one 256-thread block per edge.
__global__ __launch_bounds__(256)
void hge_scatter(const float* __restrict__ msg, const int* __restrict__ src,
                 const int* __restrict__ dst, const float* __restrict__ w,
                 float* __restrict__ out, int E) {
  int e = blockIdx.x;
  if (e >= E) return;
  int s = src[e];
  int d = dst[e];
  float we = w[e];
  float v = msg[(size_t)s * H_DIM + threadIdx.x] * we;
  __hip_atomic_fetch_add(&out[(size_t)d * H_DIM + threadIdx.x], v,
                         __ATOMIC_RELAXED, __HIP_MEMORY_SCOPE_AGENT);
}

__global__ __launch_bounds__(256)
void hge_elu(float* __restrict__ x, long n) {
  long i = (long)blockIdx.x * 256 + threadIdx.x;
  if (i < n) {
    float v = x[i];
    x[i] = v > 0.f ? v : (__expf(v) - 1.f);
  }
}

extern "C" void kernel_launch(void* const* d_in, const int* in_sizes, int n_in,
                              void* d_out, int out_size, void* d_ws, size_t ws_size,
                              hipStream_t stream) {
  const int NP = 100000, NA = 20000, FP = 512, DA = 128;
  const int EW = 200000, EC = 400000;

  const float* paper  = (const float*)d_in[0];
  const float* author = (const float*)d_in[1];
  const int*   wsrc   = (const int*)d_in[2];
  const int*   wdst   = (const int*)d_in[3];
  const float* ww     = (const float*)d_in[4];
  const int*   csrc   = (const int*)d_in[5];
  const int*   cdst   = (const int*)d_in[6];
  const float* cw     = (const float*)d_in[7];
  const float* W1sp = (const float*)d_in[8],  *b1sp = (const float*)d_in[9];
  const float* W1sa = (const float*)d_in[10], *b1sa = (const float*)d_in[11];
  const float* W1rw = (const float*)d_in[12], *b1rw = (const float*)d_in[13];
  const float* W1rc = (const float*)d_in[14], *b1rc = (const float*)d_in[15];
  const float* W2sp = (const float*)d_in[16], *b2sp = (const float*)d_in[17];
  const float* W2sa = (const float*)d_in[18], *b2sa = (const float*)d_in[19];
  const float* W2rw = (const float*)d_in[20], *b2rw = (const float*)d_in[21];
  const float* W2rc = (const float*)d_in[22], *b2rc = (const float*)d_in[23];

  float*  xp1  = (float*)d_ws;                      // [NP, 256]
  float*  xa1  = xp1  + (size_t)NP * H_DIM;         // [NA, 256]
  float*  msgP = xa1  + (size_t)NA * H_DIM;         // [NP, 256]
  float*  msgA = msgP + (size_t)NP * H_DIM;         // [NA, 256]
  __bf16* wpk  = (__bf16*)(msgA + (size_t)NA * H_DIM);  // [<=512*256] bf16

  float* outP = (float*)d_out;                      // [NP, 256]
  float* outA = outP + (size_t)NP * H_DIM;          // [NA, 256]

  dim3 blk(256);
  dim3 gP((NP + 127) / 128);
  dim3 gA((NA + 127) / 128);

#define PACK(Wm, Kd) hge_pack_w<<<dim3(Kd), blk, 0, stream>>>((Wm), wpk, (Kd))

  // ---------------- layer 1 ----------------
  PACK(W1sp, FP);
  hge_gemm_bias<<<gP, blk, 0, stream>>>(paper,  wpk, b1sp, xp1,  NP, FP);
  PACK(W1sa, DA);
  hge_gemm_bias<<<gA, blk, 0, stream>>>(author, wpk, b1sa, xa1,  NA, DA);
  PACK(W1rw, DA);
  hge_gemm_bias<<<gA, blk, 0, stream>>>(author, wpk, b1rw, msgA, NA, DA);
  hge_scatter<<<dim3(EW), blk, 0, stream>>>(msgA, wsrc, wdst, ww, xp1, EW);
  PACK(W1rc, FP);
  hge_gemm_bias<<<gP, blk, 0, stream>>>(paper,  wpk, b1rc, msgP, NP, FP);
  hge_scatter<<<dim3(EC), blk, 0, stream>>>(msgP, csrc, cdst, cw, xp1, EC);
  {
    long np = (long)NP * H_DIM, na = (long)NA * H_DIM;
    hge_elu<<<dim3((unsigned)((np + 255) / 256)), blk, 0, stream>>>(xp1, np);
    hge_elu<<<dim3((unsigned)((na + 255) / 256)), blk, 0, stream>>>(xa1, na);
  }

  // ---------------- layer 2 ----------------
  PACK(W2sp, H_DIM);
  hge_gemm_bias<<<gP, blk, 0, stream>>>(xp1, wpk, b2sp, outP, NP, H_DIM);
  PACK(W2sa, H_DIM);
  hge_gemm_bias<<<gA, blk, 0, stream>>>(xa1, wpk, b2sa, outA, NA, H_DIM);
  PACK(W2rw, H_DIM);
  hge_gemm_bias<<<gA, blk, 0, stream>>>(xa1, wpk, b2rw, msgA, NA, H_DIM);
  hge_scatter<<<dim3(EW), blk, 0, stream>>>(msgA, wsrc, wdst, ww, outP, EW);
  PACK(W2rc, H_DIM);
  hge_gemm_bias<<<gP, blk, 0, stream>>>(xp1, wpk, b2rc, msgP, NP, H_DIM);
  hge_scatter<<<dim3(EC), blk, 0, stream>>>(msgP, csrc, cdst, cw, outP, EC);
#undef PACK
}
